// EnergyGNN_43078521979000
// MI455X (gfx1250) — compile-verified, matrix operand order checked by reading, not run on hardware
//
#include <hip/hip_runtime.h>
#include <math.h>

#define N_NODES 40000
#define N_EDGES 640000
#define N_GRAPH 64
#define DIN     64
#define DH      64
#define DE      16
#define HD      128
#define EF      (N_EDGES + N_NODES)   // 680000, divisible by 16 and 8

typedef float     v2f  __attribute__((ext_vector_type(2)));
typedef float     v8f  __attribute__((ext_vector_type(8)));
typedef _Float16  v16h __attribute__((ext_vector_type(16)));

#if __has_builtin(__builtin_amdgcn_wmma_f32_16x16x4_f32)
#define HAVE_WMMA_F32 1
#else
#define HAVE_WMMA_F32 0
#endif

// ---------------------------------------------------------------------------
// WMMA GEMM: C[M x 128] = A[M x K] * W[K x 128], all row-major fp32.
// One wave = 16 rows x 128 cols (8 accumulator tiles). K % 4 == 0 (f32 path),
// K % 32 == 0 (f16 fallback). M % 16 == 0.
// Fragment packing per CDNA5 ISA 7.12.2:
//   A 16x4 : lane m=lane&15 holds row m; half=lane>>4 selects K pair {2h,2h+1}
//   B 4x16 : mirrored (col = lane&15, K pair by half)
//   C 16x16: vgpr r -> row r + 8*half, col = lane&15
// ---------------------------------------------------------------------------
__global__ __launch_bounds__(256) void gemm_node(const float* __restrict__ A,
                                                 const float* __restrict__ W,
                                                 float* __restrict__ C,
                                                 int M, int K) {
  const int wave = threadIdx.x >> 5;
  const int lane = threadIdx.x & 31;
  const long tile = (long)blockIdx.x * 8 + wave;
  const long row0 = tile * 16;
  if (row0 >= M) return;
  const int m    = lane & 15;
  const int half = lane >> 4;

  v8f acc[8];
#pragma unroll
  for (int t = 0; t < 8; ++t) acc[t] = (v8f)(0.0f);

  const float* arow = A + (row0 + m) * (long)K;

#if HAVE_WMMA_F32
  const int kq = half * 2;
  for (int k = 0; k < K; k += 4) {
    v2f a;
    a.x = arow[k + kq];
    a.y = arow[k + kq + 1];
#pragma unroll
    for (int t = 0; t < 8; ++t) {
      v2f b;
      b.x = W[(long)(k + kq)     * HD + t * 16 + m];
      b.y = W[(long)(k + kq + 1) * HD + t * 16 + m];
      acc[t] = __builtin_amdgcn_wmma_f32_16x16x4_f32(
          false, a, false, b, (short)0, acc[t], false, false);
    }
  }
#else
  for (int kb = 0; kb < K; kb += 32) {
    v16h a;
#pragma unroll
    for (int i = 0; i < 16; ++i) {
      int kl = (i < 8) ? (8 * half + i) : (16 + 8 * half + (i - 8));
      a[i] = (_Float16)arow[kb + kl];
    }
#pragma unroll
    for (int t = 0; t < 8; ++t) {
      v16h b;
#pragma unroll
      for (int i = 0; i < 16; ++i) {
        int kl = (i < 8) ? (8 * half + i) : (16 + 8 * half + (i - 8));
        b[i] = (_Float16)W[(long)(kb + kl) * HD + t * 16 + m];
      }
      acc[t] = __builtin_amdgcn_wmma_f32_16x16x32_f16(
          false, a, false, b, (short)0, acc[t], false, false);
    }
  }
#endif

  const int rbase = half * 8;
#pragma unroll
  for (int t = 0; t < 8; ++t)
#pragma unroll
    for (int r = 0; r < 8; ++r)
      C[(row0 + rbase + r) * (long)HD + t * 16 + m] = acc[t][r];
}

// ---------------------------------------------------------------------------
// Order-preserving float <-> uint for atomic max
// ---------------------------------------------------------------------------
__device__ __forceinline__ unsigned f2ord(float f) {
  unsigned u = __float_as_uint(f);
  return (u & 0x80000000u) ? ~u : (u | 0x80000000u);
}
__device__ __forceinline__ float ord2f(unsigned u) {
  return (u & 0x80000000u) ? __uint_as_float(u & 0x7fffffffu)
                           : __uint_as_float(~u);
}

// ---------------------------------------------------------------------------
// Fills / init
// ---------------------------------------------------------------------------
__global__ void fill_f32(float* p, float v, long n) {
  long i = (long)blockIdx.x * blockDim.x + threadIdx.x;
  if (i < n) p[i] = v;
}
__global__ void fill_u32(unsigned* p, unsigned v, long n) {
  long i = (long)blockIdx.x * blockDim.x + threadIdx.x;
  if (i < n) p[i] = v;
}
__global__ void init_bias(float* OUT, const float* __restrict__ cb) {
  long i = (long)blockIdx.x * blockDim.x + threadIdx.x;
  if (i < (long)N_NODES * HD) OUT[i] = cb[i & (HD - 1)];
}

// ---------------------------------------------------------------------------
// Self-loop edge attr = segment_sum(edge_attr, dst) / max(deg,1)
// ---------------------------------------------------------------------------
__global__ void loop_accum(const int* __restrict__ ei,
                           const float* __restrict__ eattr,
                           float* LOOP, float* DEG) {
  long i = (long)blockIdx.x * blockDim.x + threadIdx.x;
  if (i >= (long)N_EDGES * DE) return;
  long e = i >> 4;
  int  k = (int)(i & 15);
  int  d = ei[N_EDGES + e];
  atomicAdd(&LOOP[(long)d * DE + k], eattr[i]);
  if (k == 0) atomicAdd(&DEG[d], 1.0f);
}
__global__ void loop_norm(float* LOOP, const float* __restrict__ DEG) {
  long i = (long)blockIdx.x * blockDim.x + threadIdx.x;
  if (i >= (long)N_NODES * DE) return;
  float dg = DEG[i >> 4];
  LOOP[i] /= fmaxf(dg, 1.0f);
}

// ---------------------------------------------------------------------------
// Edge scores: wave per edge, 4 channels per lane.
// s[e,h] = sum_c leaky(xl[src]+xr[dst]+ea@we, 0.2) * att ; atomic segment max.
// ---------------------------------------------------------------------------
__global__ __launch_bounds__(256) void edge_score(
    const int* __restrict__ ei, const float* __restrict__ eattr,
    const float* __restrict__ loopat,
    const float* __restrict__ XL, const float* __restrict__ XR,
    const float* __restrict__ we, const float* __restrict__ att,
    float* __restrict__ S, unsigned* __restrict__ SMAX) {
  __shared__ float lw[DE * HD];
  __shared__ float la[HD];
  for (int i = threadIdx.x; i < DE * HD; i += 256) lw[i] = we[i];
  if (threadIdx.x < HD) la[threadIdx.x] = att[threadIdx.x];
  __syncthreads();

  const int lane = threadIdx.x & 31;
  const int wave = threadIdx.x >> 5;
  long e = (long)blockIdx.x * 8 + wave;
  if (e >= EF) return;

  int s, d;
  const float* ea;
  if (e < N_EDGES) {
    s = ei[e]; d = ei[N_EDGES + e];
    ea = eattr + e * (long)DE;
  } else {
    s = d = (int)(e - N_EDGES);
    ea = loopat + (long)s * DE;
  }

  const int c0 = lane * 4;
  float4 xlv = *(const float4*)(XL + (long)s * HD + c0);
  float4 xrv = *(const float4*)(XR + (long)d * HD + c0);

  float eev[4] = {0.f, 0.f, 0.f, 0.f};
#pragma unroll
  for (int k = 0; k < DE; ++k) {
    float eak = ea[k];
    const float* wr = &lw[k * HD + c0];
    eev[0] += eak * wr[0];
    eev[1] += eak * wr[1];
    eev[2] += eak * wr[2];
    eev[3] += eak * wr[3];
  }

  float mv[4] = {xlv.x + xrv.x + eev[0], xlv.y + xrv.y + eev[1],
                 xlv.z + xrv.z + eev[2], xlv.w + xrv.w + eev[3]};
  float sp = 0.f;
#pragma unroll
  for (int j = 0; j < 4; ++j) {
    float m  = mv[j];
    float lm = (m > 0.f) ? m : 0.2f * m;
    sp += lm * la[c0 + j];
  }
  // reduce within each 16-lane half (head 0: lanes 0-15, head 1: lanes 16-31)
#pragma unroll
  for (int off = 8; off >= 1; off >>= 1) sp += __shfl_xor(sp, off, 16);

  if ((lane & 15) == 0) {
    int h = lane >> 4;
    S[e * 2 + h] = sp;
    atomicMax(&SMAX[(long)d * 2 + h], f2ord(sp));
  }
}

// ---------------------------------------------------------------------------
// ex = exp(s - smax[dst]); den[dst] += ex  (S overwritten with ex)
// ---------------------------------------------------------------------------
__global__ void edge_exp(const int* __restrict__ ei, float* S,
                         const unsigned* __restrict__ SMAX, float* DEN) {
  long i = (long)blockIdx.x * blockDim.x + threadIdx.x;
  if (i >= (long)EF * 2) return;
  long e = i >> 1;
  int  h = (int)(i & 1);
  int  d = (e < N_EDGES) ? ei[N_EDGES + e] : (int)(e - N_EDGES);
  float sm = ord2f(SMAX[(long)d * 2 + h]);
  float ex = __expf(S[i] - sm);
  S[i] = ex;
  atomicAdd(&DEN[(long)d * 2 + h], ex);
}

// ---------------------------------------------------------------------------
// out[dst] += alpha * xl[src]  (wave per edge, 4 channels per lane)
// ---------------------------------------------------------------------------
__global__ __launch_bounds__(256) void edge_aggr(
    const int* __restrict__ ei, const float* __restrict__ S,
    const float* __restrict__ DEN, const float* __restrict__ XL,
    float* __restrict__ OUT) {
  const int lane = threadIdx.x & 31;
  const int wave = threadIdx.x >> 5;
  long e = (long)blockIdx.x * 8 + wave;
  if (e >= EF) return;
  int s, d;
  if (e < N_EDGES) { s = ei[e]; d = ei[N_EDGES + e]; }
  else             { s = d = (int)(e - N_EDGES); }
  int   h     = lane >> 4;
  float alpha = S[e * 2 + h] / (DEN[(long)d * 2 + h] + 1e-16f);
  int   c0    = lane * 4;
  float4 xlv = *(const float4*)(XL + (long)s * HD + c0);
  float* o = OUT + (long)d * HD + c0;
  atomicAdd(o + 0, alpha * xlv.x);
  atomicAdd(o + 1, alpha * xlv.y);
  atomicAdd(o + 2, alpha * xlv.z);
  atomicAdd(o + 3, alpha * xlv.w);
}

// ---------------------------------------------------------------------------
// BatchNorm stats (per-channel sum / sumsq with per-block partials)
// ---------------------------------------------------------------------------
__global__ __launch_bounds__(128) void bn_stats(const float* __restrict__ X,
                                                float* SUM, float* SSQ) {
  int  c  = threadIdx.x;
  long r0 = (long)blockIdx.x * 128;
  float s = 0.f, q = 0.f;
  for (int r = 0; r < 128; ++r) {
    long row = r0 + r;
    if (row >= N_NODES) break;
    float v = X[row * HD + c];
    s += v;
    q += v * v;
  }
  atomicAdd(&SUM[c], s);
  atomicAdd(&SSQ[c], q);
}

__global__ void bn_act(const float* __restrict__ X,
                       const float* __restrict__ SUM,
                       const float* __restrict__ SSQ,
                       const float* __restrict__ g,
                       const float* __restrict__ b,
                       float* __restrict__ Y) {
  long i = (long)blockIdx.x * blockDim.x + threadIdx.x;
  if (i >= (long)N_NODES * HD) return;
  int   c   = (int)(i & (HD - 1));
  float mu  = SUM[c] * (1.0f / N_NODES);
  float var = SSQ[c] * (1.0f / N_NODES) - mu * mu;
  float v   = (X[i] - mu) * rsqrtf(var + 1e-5f) * g[c] + b[c];
  Y[i] = (v > 0.f) ? v : 0.01f * v;
}

// ---------------------------------------------------------------------------
// Global mean pool + MLP head
// ---------------------------------------------------------------------------
__global__ __launch_bounds__(128) void pool_accum(const float* __restrict__ X,
                                                  const int* __restrict__ batch,
                                                  float* POOL, float* CNT) {
  int n = blockIdx.x;
  int c = threadIdx.x;
  int g = batch[n];
  atomicAdd(&POOL[(long)g * HD + c], X[(long)n * HD + c]);
  if (c == 0) atomicAdd(&CNT[g], 1.0f);
}

__global__ __launch_bounds__(64) void head(
    const float* __restrict__ POOL, const float* __restrict__ CNT,
    const float* __restrict__ fc1w, const float* __restrict__ fc1b,
    const float* __restrict__ fc2w, const float* __restrict__ fc2b,
    float* __restrict__ out) {
  __shared__ float red[64];
  int g = blockIdx.x, t = threadIdx.x;
  float cinv = 1.0f / fmaxf(CNT[g], 1.0f);
  float acc  = fc1b[t];
  for (int c = 0; c < HD; ++c)
    acc += POOL[(long)g * HD + c] * cinv * fc1w[c * DH + t];
  acc = fmaxf(acc, 0.0f);
  red[t] = acc * fc2w[t];
  __syncthreads();
  for (int off = 32; off >= 1; off >>= 1) {
    if (t < off) red[t] += red[t + off];
    __syncthreads();
  }
  if (t == 0) out[g] = red[0] + fc2b[0];
}

// ---------------------------------------------------------------------------
// Launch
// ---------------------------------------------------------------------------
extern "C" void kernel_launch(void* const* d_in, const int* in_sizes, int n_in,
                              void* d_out, int out_size, void* d_ws, size_t ws_size,
                              hipStream_t stream) {
  (void)in_sizes; (void)n_in; (void)out_size; (void)ws_size;

  const float* x     = (const float*)d_in[0];
  const int*   ei    = (const int*)d_in[1];
  const float* eattr = (const float*)d_in[2];
  const int*   batch = (const int*)d_in[3];
  const float* fc1w  = (const float*)d_in[25];
  const float* fc1b  = (const float*)d_in[26];
  const float* fc2w  = (const float*)d_in[27];
  const float* fc2b  = (const float*)d_in[28];

  char*  ws  = (char*)d_ws;
  size_t off = 0;
  auto carve = [&](size_t bytes) {
    char* p = ws + off;
    off += (bytes + 255) & ~(size_t)255;
    return p;
  };
  float*    F0   = (float*)carve((size_t)N_NODES * HD * 4);
  float*    F1   = (float*)carve((size_t)N_NODES * HD * 4);
  float*    XL   = (float*)carve((size_t)N_NODES * HD * 4);
  float*    XR   = (float*)carve((size_t)N_NODES * HD * 4);
  float*    LOOP = (float*)carve((size_t)N_NODES * DE * 4);
  float*    DEG  = (float*)carve((size_t)N_NODES * 4);
  float*    S    = (float*)carve((size_t)EF * 2 * 4);
  unsigned* SMAX = (unsigned*)carve((size_t)N_NODES * 2 * 4);
  float*    DEN  = (float*)carve((size_t)N_NODES * 2 * 4);
  float*    BNS  = (float*)carve(256 * 4);          // [0..127]=sum, [128..255]=sumsq
  float*    POOL = (float*)carve((size_t)N_GRAPH * HD * 4);
  float*    CNT  = (float*)carve((size_t)N_GRAPH * 4);

  const int TB = 256;
  auto blk = [](long n, int t) { return (unsigned)((n + t - 1) / t); };

  // self-loop attributes (segment mean of incoming edge attrs)
  fill_f32<<<blk((long)N_NODES * DE, TB), TB, 0, stream>>>(LOOP, 0.f, (long)N_NODES * DE);
  fill_f32<<<blk(N_NODES, TB), TB, 0, stream>>>(DEG, 0.f, N_NODES);
  loop_accum<<<blk((long)N_EDGES * DE, TB), TB, 0, stream>>>(ei, eattr, LOOP, DEG);
  loop_norm<<<blk((long)N_NODES * DE, TB), TB, 0, stream>>>(LOOP, DEG);

  const float* cur  = x;
  int          curK = DIN;
  const unsigned gemm_blocks = blk((long)(N_NODES / 16), 8);  // 8 waves/block
  const unsigned edge_blocks = (unsigned)(EF / 8);            // wave per edge

  for (int l = 0; l < 3; ++l) {
    const float* wl = (const float*)d_in[4 + l * 7 + 0];
    const float* wr = (const float*)d_in[4 + l * 7 + 1];
    const float* we = (const float*)d_in[4 + l * 7 + 2];
    const float* at = (const float*)d_in[4 + l * 7 + 3];
    const float* cb = (const float*)d_in[4 + l * 7 + 4];
    const float* gg = (const float*)d_in[4 + l * 7 + 5];
    const float* bb = (const float*)d_in[4 + l * 7 + 6];

    gemm_node<<<gemm_blocks, TB, 0, stream>>>(cur, wl, XL, N_NODES, curK);
    gemm_node<<<gemm_blocks, TB, 0, stream>>>(cur, wr, XR, N_NODES, curK);

    fill_u32<<<blk((long)N_NODES * 2, TB), TB, 0, stream>>>(SMAX, 0u, (long)N_NODES * 2);
    fill_f32<<<blk((long)N_NODES * 2, TB), TB, 0, stream>>>(DEN, 0.f, (long)N_NODES * 2);

    edge_score<<<edge_blocks, TB, 0, stream>>>(ei, eattr, LOOP, XL, XR, we, at, S, SMAX);
    edge_exp<<<blk((long)EF * 2, TB), TB, 0, stream>>>(ei, S, SMAX, DEN);

    init_bias<<<blk((long)N_NODES * HD, TB), TB, 0, stream>>>(F1, cb);
    edge_aggr<<<edge_blocks, TB, 0, stream>>>(ei, S, DEN, XL, F1);

    fill_f32<<<1, 256, 0, stream>>>(BNS, 0.f, 256);
    bn_stats<<<blk(N_NODES, 128), 128, 0, stream>>>(F1, BNS, BNS + 128);
    bn_act<<<blk((long)N_NODES * HD, TB), TB, 0, stream>>>(F1, BNS, BNS + 128, gg, bb, F0);

    cur  = F0;
    curK = HD;
  }

  fill_f32<<<blk((long)N_GRAPH * HD, TB), TB, 0, stream>>>(POOL, 0.f, (long)N_GRAPH * HD);
  fill_f32<<<1, N_GRAPH, 0, stream>>>(CNT, 0.f, N_GRAPH);
  pool_accum<<<N_NODES, 128, 0, stream>>>(F0, batch, POOL, CNT);
  head<<<N_GRAPH, 64, 0, stream>>>(POOL, CNT, fc1w, fc1b, fc2w, fc2b, (float*)d_out);
}
